// LSTM_CRF_32143535243634
// MI455X (gfx1250) — compile-verified, matrix operand order checked by reading, not run on hardware
//
#include <hip/hip_runtime.h>
#include <hip/hip_bf16.h>
#include <stdint.h>

#define D      256
#define H      256
#define LTAGS  9
#define BATCH  64
#define SEQ    512
#define G4H    1024           // 4*H
#define NTOK   (SEQ * BATCH)  // 32768

typedef __bf16 bf16_t;
typedef __attribute__((ext_vector_type(16))) __bf16 v16bf;
typedef __attribute__((ext_vector_type(8)))  __bf16 v8bf;
typedef __attribute__((ext_vector_type(8)))  float  v8f;
typedef __attribute__((ext_vector_type(4)))  int    v4i;

#if __has_builtin(__builtin_amdgcn_global_load_async_to_lds_b128) && \
    __has_builtin(__builtin_amdgcn_s_wait_asynccnt)
#define ASYNC_LDS 1
#endif

__device__ __forceinline__ bf16_t f2bf(float f) { return (bf16_t)f; }
__device__ __forceinline__ float  bf2f(bf16_t b) { return (float)b; }
__device__ __forceinline__ float  sigf(float x) { return 1.0f / (1.0f + __expf(-x)); }

__device__ __forceinline__ v8f wmma_bf16(v16bf a, v16bf b, v8f c) {
    return __builtin_amdgcn_wmma_f32_16x16x32_bf16(false, a, false, b, (short)0, c, false, false);
}

__device__ __forceinline__ v16bf cat8(v8bf lo, v8bf hi) {
    return __builtin_shufflevector(lo, hi, 0,1,2,3,4,5,6,7,8,9,10,11,12,13,14,15);
}

// one 16-byte global -> LDS copy (ASYNCcnt-tracked when available)
__device__ __forceinline__ void async_cp16(const void* g, void* l) {
#ifdef ASYNC_LDS
    __builtin_amdgcn_global_load_async_to_lds_b128((v4i*)g, (v4i*)l, 0, 0);
#else
    *(uint4*)l = *(const uint4*)g;
#endif
}
__device__ __forceinline__ void wait_async0() {
#ifdef ASYNC_LDS
    __builtin_amdgcn_s_wait_asynccnt(0);
#endif
}

// ---------------------------------------------------------------------------
// Kernel 0: pack a (4H x 256) f32 weight matrix into WMMA B-fragment bf16
// layout (ISA 16-bit B layout: lanes 0-15 K 0-7/16-23, lanes 16-31 K 8-15/
// 24-31).  One 32B coalesced load per lane at consume time.
// ---------------------------------------------------------------------------
__global__ void pack_weights(const float* __restrict__ W, bf16_t* __restrict__ P) {
    int e = blockIdx.x * blockDim.x + threadIdx.x;   // [0, 4H*H)
    if (e >= G4H * H) return;
    int i     = e & 15;
    int lane  = (e >> 4) & 31;
    int kc    = (e >> 9) & 7;
    int ntile = e >> 12;                             // [0, 64)
    int n     = ntile * 16 + (lane & 15);            // row of W (output unit)
    int kb    = (lane < 16) ? 0 : 8;
    int k     = kc * 32 + ((i < 8) ? (kb + i) : (16 + kb + (i - 8)));
    P[e] = f2bf(W[(size_t)n * H + k]);
}

// Tiny init: zero the cross-block sync flags (must run before lstm_scan).
__global__ void zero_flags(int* __restrict__ flags) {
    if (threadIdx.x < 8) flags[threadIdx.x] = 0;
}

// ---------------------------------------------------------------------------
// Kernel 1: X[t,b,:] = Emb[x[b,t]] @ Wih^T + bias   (WMMA, bf16 out)
// High-occupancy throughput kernel; L2 latency hidden by parallel waves.
// ---------------------------------------------------------------------------
__global__ __launch_bounds__(256, 1)
void input_gemm(const int* __restrict__ x, const float* __restrict__ E,
                const bf16_t* __restrict__ WihPk, const float* __restrict__ bias,
                bf16_t* __restrict__ X)
{
    __shared__ bf16_t elds[16 * D];                  // 8 KB
    const int tid   = threadIdx.x;
    const int mtile = blockIdx.x;

    {   // gather + convert 16 embedding rows
        int tok  = tid >> 4;
        int part = tid & 15;
        int token = mtile * 16 + tok;                // token = t*64 + b
        int t = token >> 6, b_ = token & 63;
        int id = x[b_ * SEQ + t];
        const float* er = E + (size_t)id * D;
        #pragma unroll
        for (int q = 0; q < 16; ++q) {
            int k = part * 16 + q;
            elds[tok * D + k] = f2bf(er[k]);
        }
    }
    __syncthreads();

    const int w = tid >> 5, lane = tid & 31;
    const int l15 = lane & 15;
    const int kb  = (lane < 16) ? 0 : 8;
    const int rh  = (lane < 16) ? 0 : 8;

    v16bf afrag[8];
    #pragma unroll
    for (int kc = 0; kc < 8; ++kc) {
        v8bf lo = *(const v8bf*)&elds[l15 * D + kc * 32 + kb];
        v8bf hi = *(const v8bf*)&elds[l15 * D + kc * 32 + 16 + kb];
        afrag[kc] = cat8(lo, hi);
    }

    for (int nn = 0; nn < 8; ++nn) {
        int ntile = w * 8 + nn;
        int col   = ntile * 16 + l15;
        float bv  = bias[col];
        v8f acc;
        #pragma unroll
        for (int r = 0; r < 8; ++r) acc[r] = bv;
        #pragma unroll
        for (int kc = 0; kc < 8; ++kc) {
            v16bf bfrag = *(const v16bf*)&WihPk[((size_t)(ntile * 8 + kc) * 32 + lane) * 16];
            acc = wmma_bf16(afrag[kc], bfrag, acc);
        }
        #pragma unroll
        for (int r = 0; r < 8; ++r) {
            int token = mtile * 16 + r + rh;
            X[(size_t)token * G4H + col] = f2bf(acc[r]);
        }
    }
}

// ---------------------------------------------------------------------------
// Kernel 2: cooperative persistent LSTM scan.  Grid = 4 blocks x 1024 thr:
// dir = bid&1 (fwd/bwd), p = bid>>1 (unit-half).  Each block owns hidden
// units [p*128, p*128+128): its 32 ntiles of packed Whh (256 KB bf16) are
// async-staged into LDS ONCE, so every per-step B fragment is an LDS read —
// no L2 latency on the serial path.  Full h (64x256 bf16, 32 KB) is kept in
// LDS; per step each block computes its half of h (32 WMMAs/wave), then the
// halves are exchanged through L2 with acquire/release flag handshakes
// (monotonic step counters, double-buffered 16 KB slabs, ack flags), and the
// partner half is pulled back into LDS with async-to-LDS loads.
// ---------------------------------------------------------------------------
__global__ __launch_bounds__(1024, 1)
void lstm_scan(const bf16_t* __restrict__ Xf, const bf16_t* __restrict__ Xb,
               const bf16_t* __restrict__ PkF, const bf16_t* __restrict__ PkB,
               float* __restrict__ hfOut, float* __restrict__ hbOut,
               bf16_t* __restrict__ xchg,   // [dir][p][slab] 64x128 bf16 each
               int* __restrict__ flags)     // [0..3]=prod(dir,p) [4..7]=cons(dir,p)
{
    __shared__ bf16_t wl[32 * 8 * 512];              // 256 KB: this half's Whh frags
    __shared__ bf16_t hl[BATCH * H];                 // 32 KB: full h state
    const int bid = blockIdx.x;
    const int dir = bid & 1;
    const int p   = bid >> 1;                        // unit-half 0/1
    const bf16_t* __restrict__ X     = dir ? Xb  : Xf;
    const bf16_t* __restrict__ WhhPk = dir ? PkB : PkF;
    float*        __restrict__ hs    = dir ? hbOut : hfOut;

    const int tid = threadIdx.x;

    // prologue: async-stage 4 x 64KB weight chunks (one per gate) into LDS.
    // chunk for gate: frags (gate*16 + p*8 + q)*8 + kc, q/kc nested -> contiguous.
    #pragma unroll
    for (int gate = 0; gate < 4; ++gate) {
        const char* src = (const char*)WhhPk + ((size_t)(gate * 16 + p * 8) * 8) * 1024;
        char* dst = (char*)wl + (size_t)gate * 65536;
        #pragma unroll
        for (int i = 0; i < 4; ++i)
            async_cp16(src + tid * 16 + i * 16384, dst + tid * 16 + i * 16384);
    }
    {   // zero h
        uint4 z = {0, 0, 0, 0};
        *(uint4*)((char*)hl + tid * 32)      = z;
        *(uint4*)((char*)hl + tid * 32 + 16) = z;
    }
    wait_async0();
    __syncthreads();                                 // wl + hl ready

    const int w = tid >> 5, lane = tid & 31;
    const int l15 = lane & 15;
    const int kb  = (lane < 16) ? 0 : 8;
    const int rh  = (lane < 16) ? 0 : 8;
    const int m   = w & 3;                           // batch tile 0..3
    const int u   = w >> 2;                          // local unit tile 0..7
    const int gcol = p * 128 + u * 16 + l15;         // global h column

    const int prodMe = dir * 2 + p,     prodPt = dir * 2 + (1 - p);
    const int consMe = 4 + dir * 2 + p, consPt = 4 + dir * 2 + (1 - p);
    bf16_t* slabMe = xchg + (size_t)((dir * 2 + p) * 2) * 8192;
    const bf16_t* slabPt = xchg + (size_t)((dir * 2 + (1 - p)) * 2) * 8192;

    v8f cst;
    #pragma unroll
    for (int r = 0; r < 8; ++r) cst[r] = 0.0f;

    for (int step = 0; step < SEQ; ++step) {
        const int t = dir ? (SEQ - 1 - step) : step;

        // ---- A: compute g = X[t] + h @ Whh^T for this block's 128 units ----
        v16bf afrag[8];
        const int arow = m * 16 + l15;
        #pragma unroll
        for (int kc = 0; kc < 8; ++kc) {
            v8bf lo = *(const v8bf*)&hl[arow * H + kc * 32 + kb];
            v8bf hi = *(const v8bf*)&hl[arow * H + kc * 32 + 16 + kb];
            afrag[kc] = cat8(lo, hi);
        }
        v8f acc[4];
        #pragma unroll
        for (int gate = 0; gate < 4; ++gate) {
            int col = gate * 256 + gcol;
            const bf16_t* xp = X + (size_t)t * BATCH * G4H + col;
            #pragma unroll
            for (int r = 0; r < 8; ++r)
                acc[gate][r] = bf2f(xp[(size_t)(m * 16 + r + rh) * G4H]);
        }
        #pragma unroll
        for (int kc = 0; kc < 8; ++kc) {
            #pragma unroll
            for (int gate = 0; gate < 4; ++gate) {
                v16bf bfrag = *(const v16bf*)&wl[(size_t)(((gate * 8 + u) * 8) + kc) * 512 + lane * 16];
                acc[gate] = wmma_bf16(afrag[kc], bfrag, acc[gate]);
            }
        }
        float hv[8];
        #pragma unroll
        for (int r = 0; r < 8; ++r) {
            float iv = sigf(acc[0][r]);
            float fv = sigf(acc[1][r]);
            float gv = tanhf(acc[2][r]);
            float ov = sigf(acc[3][r]);
            float c  = fv * cst[r] + iv * gv;
            cst[r] = c;
            hv[r]  = ov * tanhf(c);
        }

        // ---- B: old h consumed; make sure my outgoing slab is free --------
        __syncthreads();
        if (tid == 0 && step >= 2) {
            while (__hip_atomic_load(&flags[consPt], __ATOMIC_ACQUIRE,
                                     __HIP_MEMORY_SCOPE_AGENT) < step - 1)
                __builtin_amdgcn_s_sleep(1);
        }
        __syncthreads();

        // ---- C: publish my half (LDS h, global hs, exchange slab) ---------
        bf16_t* slab = slabMe + (size_t)(step & 1) * 8192;
        #pragma unroll
        for (int r = 0; r < 8; ++r) {
            int row = m * 16 + r + rh;
            bf16_t hb16 = f2bf(hv[r]);
            hl[row * H + gcol] = hb16;
            slab[row * 128 + u * 16 + l15] = hb16;
            hs[((size_t)t * BATCH + row) * H + gcol] = hv[r];
        }
        __threadfence();                             // device-visible
        __syncthreads();

        // ---- D: handshake ------------------------------------------------
        if (tid == 0) {
            __hip_atomic_store(&flags[prodMe], step + 1, __ATOMIC_RELEASE,
                               __HIP_MEMORY_SCOPE_AGENT);
            while (__hip_atomic_load(&flags[prodPt], __ATOMIC_ACQUIRE,
                                     __HIP_MEMORY_SCOPE_AGENT) < step + 1)
                __builtin_amdgcn_s_sleep(1);
        }
        __syncthreads();

        // ---- F: pull partner half into LDS (async), ack --------------------
        {
            const char* ps = (const char*)(slabPt + (size_t)(step & 1) * 8192);
            int o   = tid * 16;                      // 16 B per thread (16 KB total)
            int row = o >> 8;                        // 256 B per slab row
            int rem = o & 255;
            async_cp16(ps + o, (char*)hl + row * 512 + (1 - p) * 256 + rem);
        }
        wait_async0();
        __syncthreads();                             // full h ready for next step
        if (tid == 0) {
            __hip_atomic_store(&flags[consMe], step + 1, __ATOMIC_RELEASE,
                               __HIP_MEMORY_SCOPE_AGENT);
        }
    }
}

// ---------------------------------------------------------------------------
// Kernel 3: em = [hf | hb] @ Wo + bo   (N=9: wave-per-token VALU + shfl)
// ---------------------------------------------------------------------------
__global__ __launch_bounds__(256, 1)
void emissions(const float* __restrict__ hf, const float* __restrict__ hb,
               const float* __restrict__ Wo, const float* __restrict__ bo,
               float* __restrict__ em)
{
    __shared__ float wl[2 * H * LTAGS];              // 18 KB
    __shared__ float bl[LTAGS];
    int tid = threadIdx.x;
    for (int i = tid; i < 2 * H * LTAGS; i += 256) wl[i] = Wo[i];
    if (tid < LTAGS) bl[tid] = bo[tid];
    __syncthreads();

    int w = tid >> 5, lane = tid & 31;
    int token = blockIdx.x * 8 + w;
    const float* hfr = hf + (size_t)token * H;
    const float* hbr = hb + (size_t)token * H;

    float acc[LTAGS];
    #pragma unroll
    for (int l = 0; l < LTAGS; ++l) acc[l] = 0.0f;
    #pragma unroll
    for (int kk = 0; kk < 8; ++kk) {
        int k = kk * 32 + lane;
        float a = hfr[k], b2 = hbr[k];
        #pragma unroll
        for (int l = 0; l < LTAGS; ++l)
            acc[l] += a * wl[k * LTAGS + l] + b2 * wl[(H + k) * LTAGS + l];
    }
    #pragma unroll
    for (int off = 16; off > 0; off >>= 1) {
        #pragma unroll
        for (int l = 0; l < LTAGS; ++l)
            acc[l] += __shfl_xor(acc[l], off, 32);
    }
    if (lane == 0) {
        int b = token & 63, t = token >> 6;
        #pragma unroll
        for (int l = 0; l < LTAGS; ++l)
            em[(size_t)b * SEQ * LTAGS + t * LTAGS + l] = acc[l] + bl[l];
    }
}

// ---------------------------------------------------------------------------
// Kernel 4: CRF logZ (forward algorithm), gold score, mean(logZ - gold).
// ---------------------------------------------------------------------------
__global__ __launch_bounds__(64, 1)
void crf(const float* __restrict__ em, const int* __restrict__ x, const int* __restrict__ y,
         const float* __restrict__ T, const float* __restrict__ startv,
         const float* __restrict__ endv, float* __restrict__ out)
{
    __shared__ float Ts[LTAGS * LTAGS], ss[LTAGS], es[LTAGS], diff[BATCH];
    int tid = threadIdx.x;
    if (tid < LTAGS * LTAGS) Ts[tid] = T[tid];
    if (tid < LTAGS) { ss[tid] = startv[tid]; es[tid] = endv[tid]; }
    __syncthreads();

    const int b = tid;
    const float* emb_ = em + (size_t)b * SEQ * LTAGS;
    const int* xb = x + b * SEQ;
    const int* yb = y + b * SEQ;

    float alpha[LTAGS];
    #pragma unroll
    for (int j = 0; j < LTAGS; ++j) alpha[j] = ss[j] + emb_[j];

    for (int t = 1; t < SEQ; ++t) {
        if (xb[t] != 0) {
            float na[LTAGS];
            #pragma unroll
            for (int j = 0; j < LTAGS; ++j) {
                float mx = -1e30f;
                #pragma unroll
                for (int i = 0; i < LTAGS; ++i) mx = fmaxf(mx, alpha[i] + Ts[i * LTAGS + j]);
                float s = 0.0f;
                #pragma unroll
                for (int i = 0; i < LTAGS; ++i) s += __expf(alpha[i] + Ts[i * LTAGS + j] - mx);
                na[j] = mx + __logf(s) + emb_[t * LTAGS + j];
            }
            #pragma unroll
            for (int j = 0; j < LTAGS; ++j) alpha[j] = na[j];
        }
    }
    float mx = -1e30f;
    #pragma unroll
    for (int j = 0; j < LTAGS; ++j) mx = fmaxf(mx, alpha[j] + es[j]);
    float s = 0.0f;
    #pragma unroll
    for (int j = 0; j < LTAGS; ++j) s += __expf(alpha[j] + es[j] - mx);
    float logZ = mx + __logf(s);

    int y0 = yb[0];
    float score = ss[y0] + emb_[y0];
    int cnt = (xb[0] != 0) ? 1 : 0;
    int prev = y0;
    for (int t = 1; t < SEQ; ++t) {
        int yt = yb[t];
        if (xb[t] != 0) { score += Ts[prev * LTAGS + yt] + emb_[t * LTAGS + yt]; cnt++; }
        prev = yt;
    }
    int lastIdx = cnt - 1; if (lastIdx < 0) lastIdx = 0;
    score += es[yb[lastIdx]];

    diff[b] = logZ - score;
    __syncthreads();
    if (tid == 0) {
        float s2 = 0.0f;
        for (int i = 0; i < BATCH; ++i) s2 += diff[i];
        out[0] = s2 / (float)BATCH;
    }
}

// ---------------------------------------------------------------------------
extern "C" void kernel_launch(void* const* d_in, const int* in_sizes, int n_in,
                              void* d_out, int out_size, void* d_ws, size_t ws_size,
                              hipStream_t stream)
{
    (void)in_sizes; (void)n_in; (void)out_size; (void)ws_size;

    const int*   x     = (const int*)  d_in[0];
    const int*   y     = (const int*)  d_in[1];
    const float* E     = (const float*)d_in[2];
    const float* Wih_f = (const float*)d_in[3];
    const float* Whh_f = (const float*)d_in[4];
    const float* b_f   = (const float*)d_in[5];
    const float* Wih_b = (const float*)d_in[6];
    const float* Whh_b = (const float*)d_in[7];
    const float* b_b   = (const float*)d_in[8];
    const float* Wo    = (const float*)d_in[9];
    const float* bo    = (const float*)d_in[10];
    const float* T     = (const float*)d_in[11];
    const float* start = (const float*)d_in[12];
    const float* endv  = (const float*)d_in[13];

    char* ws = (char*)d_ws;
    size_t off = 0;
    bf16_t* Xf    = (bf16_t*)(ws + off); off += (size_t)NTOK * G4H * 2;   // 64 MiB
    bf16_t* Xb    = (bf16_t*)(ws + off); off += (size_t)NTOK * G4H * 2;   // 64 MiB
    bf16_t* PkIhF = (bf16_t*)(ws + off); off += (size_t)G4H * H * 2;      // 512 KiB
    bf16_t* PkHhF = (bf16_t*)(ws + off); off += (size_t)G4H * H * 2;
    bf16_t* PkIhB = (bf16_t*)(ws + off); off += (size_t)G4H * H * 2;
    bf16_t* PkHhB = (bf16_t*)(ws + off); off += (size_t)G4H * H * 2;
    float*  hfBuf = (float*)(ws + off);  off += (size_t)NTOK * H * 4;     // 32 MiB
    float*  hbBuf = (float*)(ws + off);  off += (size_t)NTOK * H * 4;     // 32 MiB
    float*  emBuf = (float*)(ws + off);  off += (size_t)BATCH * SEQ * LTAGS * 4;
    bf16_t* xchg  = (bf16_t*)(ws + off); off += (size_t)8 * 8192 * 2;     // 128 KiB
    int*    flags = (int*)(ws + off);    off += 256;

    const int packBlocks = (G4H * H) / 256;          // 1024
    pack_weights<<<packBlocks, 256, 0, stream>>>(Wih_f, PkIhF);
    pack_weights<<<packBlocks, 256, 0, stream>>>(Whh_f, PkHhF);
    pack_weights<<<packBlocks, 256, 0, stream>>>(Wih_b, PkIhB);
    pack_weights<<<packBlocks, 256, 0, stream>>>(Whh_b, PkHhB);
    zero_flags<<<1, 32, 0, stream>>>(flags);

    input_gemm<<<NTOK / 16, 256, 0, stream>>>(x, E, PkIhF, b_f, Xf);
    input_gemm<<<NTOK / 16, 256, 0, stream>>>(x, E, PkIhB, b_b, Xb);

    lstm_scan<<<4, 1024, 0, stream>>>(Xf, Xb, PkHhF, PkHhB, hfBuf, hbBuf, xchg, flags);

    emissions<<<NTOK / 8, 256, 0, stream>>>(hfBuf, hbBuf, Wo, bo, emBuf);

    crf<<<1, BATCH, 0, stream>>>(emBuf, x, y, T, start, endv, (float*)d_out);
}
